// FeatureFusionPHA_59055800320142
// MI455X (gfx1250) — matrix operand order repaired and checked by reading, not compiled, not run on hardware
//
#include <hip/hip_runtime.h>
#include <hip/hip_bf16.h>

// ---------------------------------------------------------------------------
// FeatureFusionPHA for MI455X (gfx1250, wave32, WMMA + async-to-LDS).
// All heavy math runs through v_wmma_f32_16x16x32_bf16, f32 accumulate.
// GEMM-to-GEMM intermediates (p, kv, gate, a, o, td) and all weights are kept
// in memory as bf16; bf16 tiles are staged into LDS with
// GLOBAL_LOAD_ASYNC_TO_LDS_B128 (ASYNCcnt) -- no VGPR round trip.  fp32 is
// kept only where VALU consumers need it (LN, softmax, dwconv, BN fuse).
// Every WMMA fragment is two contiguous 16-byte LDS runs -> ds_load_b128.
// ---------------------------------------------------------------------------

typedef __attribute__((ext_vector_type(16))) __bf16    v16bf;
typedef __attribute__((ext_vector_type(8)))  float     v8f;
typedef __attribute__((ext_vector_type(4)))  unsigned  v4u;

struct U32x8 { v4u lo, hi; };

__device__ __forceinline__ unsigned bfbits(float f) {
    unsigned u = __builtin_bit_cast(unsigned, f);
    return (u + 0x7FFFu + ((u >> 16) & 1u)) >> 16;   // round-to-nearest-even
}
__device__ __forceinline__ unsigned packbf2(float lo, float hi) {
    return bfbits(lo) | (bfbits(hi) << 16);
}
__device__ __forceinline__ __bf16 bfval(float f) {
    unsigned short s = (unsigned short)bfbits(f);
    return __builtin_bit_cast(__bf16, s);
}
__device__ __forceinline__ float bf2f(unsigned short s) {
    return __builtin_bit_cast(float, (unsigned)s << 16);
}
// Fragment = two contiguous 16B runs in LDS -> 2x ds_load_b128, one bit_cast.
__device__ __forceinline__ v16bf load_frag(const __bf16* p0, const __bf16* p1) {
    U32x8 u;
    u.lo = *(const v4u*)p0;
    u.hi = *(const v4u*)p1;
    return __builtin_bit_cast(v16bf, u);
}
// Async 16B global -> LDS copy (GLOBAL_LOAD_ASYNC_TO_LDS_B128, ASYNCcnt).
// LDS offset = low 32 bits of the generic address of a __shared__ object.
__device__ __forceinline__ void async_copy16(const __bf16* lds_dst, const void* gsrc) {
    unsigned lds_off = (unsigned)(size_t)lds_dst;
    asm volatile("global_load_async_to_lds_b128 %0, %1, off"
                 :: "v"(lds_off), "v"(gsrc)
                 : "memory");
}
__device__ __forceinline__ void wait_async0() {
    asm volatile("s_wait_asynccnt 0" ::: "memory");
}

// one-shot fp32 -> bf16 conversion (weights)
__global__ __launch_bounds__(256)
void f32_to_bf16(const float* __restrict__ src, __bf16* __restrict__ dst, int n)
{
    int i = blockIdx.x * 256 + threadIdx.x;
    if (i < n) dst[i] = bfval(src[i]);
}

// ---------------------------------------------------------------------------
// Generic WMMA GEMM:  Y[M,Nout] = act( X[M,K] @ Wb[Nout,K]^T + bias )
// TA = float (convert while staging, DScnt) or __bf16 (async b128 copy).
// TO = float or __bf16 output.  Wb is always bf16 (pre-converted, async).
// X is split across two row-major sources (virtual concat along K):
//   k <  K1 : X1 + row*ldx1 + k        k >= K1 : X2 + row*ldx2 + (k-K1)
// Block = 256 threads (8 wave32 waves), tile 128x128, K-step 32,
// double-buffered LDS.  Wave grid 4x2; wave = 2 M-frags x 4 N-frags.
// ACT: 0 = none, 1 = relu, 2 = sigmoid.
// ---------------------------------------------------------------------------
template<int ACT, typename TA, typename TO>
__global__ __launch_bounds__(256)
void gemm_wmma(const TA* __restrict__ X1, int ldx1, int K1,
               const TA* __restrict__ X2, int ldx2, int Ktot,
               const __bf16* __restrict__ Wb, const float* __restrict__ bias,
               TO* __restrict__ Y, int Nout)
{
    // pitch 40 bf16 = 80B rows: 16B-aligned (b128 frag reads), conflict-free.
    __shared__ __bf16 As[2][128][40];   // [m][k]
    __shared__ __bf16 Bs[2][128][40];   // [n][k] (Wb tile)

    const int tid   = threadIdx.x;
    const int lane  = tid & 31;
    const int wid   = tid >> 5;
    const int waveM = wid & 3;
    const int waveN = wid >> 2;
    const int mBase = blockIdx.x * 128;
    const int nBase = blockIdx.y * 128;
    const int khalf = lane >> 4;
    const int l15   = lane & 15;

    v8f acc[2][4] = {};

    auto stage = [&](int buf, int kb) {
        // ---- A tile 128x32 ----
        if constexpr (sizeof(TA) == 2) {          // bf16: async b128 copy
            #pragma unroll
            for (int i = 0; i < 2; ++i) {
                int g  = tid + i * 256;           // 0..511 groups of 8 elems
                int r  = g >> 2;
                int c8 = (g & 3) << 3;
                int kg = kb + c8;
                const TA* ap = (kg < K1)
                    ? (X1 + (size_t)(mBase + r) * ldx1 + kg)
                    : (X2 + (size_t)(mBase + r) * ldx2 + (kg - K1));
                async_copy16(&As[buf][r][c8], ap);
            }
        } else {                                   // fp32: convert to bf16
            #pragma unroll
            for (int i = 0; i < 4; ++i) {
                int g  = tid + i * 256;            // 0..1023 float4 groups
                int r  = g >> 3;
                int c4 = (g & 7) << 2;
                int kg = kb + c4;
                const TA* ap = (kg < K1)
                    ? (X1 + (size_t)(mBase + r) * ldx1 + kg)
                    : (X2 + (size_t)(mBase + r) * ldx2 + (kg - K1));
                float4 af = *(const float4*)ap;
                *(unsigned*)&As[buf][r][c4]     = packbf2(af.x, af.y);
                *(unsigned*)&As[buf][r][c4 + 2] = packbf2(af.z, af.w);
            }
        }
        // ---- B tile 128x32: always bf16 async copy ----
        #pragma unroll
        for (int i = 0; i < 2; ++i) {
            int g  = tid + i * 256;
            int r  = g >> 2;
            int c8 = (g & 3) << 3;
            async_copy16(&Bs[buf][r][c8],
                         Wb + (size_t)(nBase + r) * Ktot + kb + c8);
        }
    };

    stage(0, 0);
    int buf = 0;
    for (int kb = 0; kb < Ktot; kb += 32) {
        wait_async0();                 // previous iteration's async prefetch
        __syncthreads();
        if (kb + 32 < Ktot) stage(buf ^ 1, kb + 32);   // prefetch next tile

        const __bf16 (*Ab)[40] = As[buf];
        const __bf16 (*Bb)[40] = Bs[buf];
        v16bf afrag[2], bfrag[4];
        #pragma unroll
        for (int fm = 0; fm < 2; ++fm) {
            int m = waveM * 32 + fm * 16 + l15;
            afrag[fm] = load_frag(&Ab[m][khalf * 8], &Ab[m][16 + khalf * 8]);
        }
        #pragma unroll
        for (int fn = 0; fn < 4; ++fn) {
            int n = waveN * 64 + fn * 16 + l15;
            bfrag[fn] = load_frag(&Bb[n][khalf * 8], &Bb[n][16 + khalf * 8]);
        }
        #pragma unroll
        for (int fm = 0; fm < 2; ++fm)
            #pragma unroll
            for (int fn = 0; fn < 4; ++fn)
                acc[fm][fn] = __builtin_amdgcn_wmma_f32_16x16x32_bf16(
                    false, afrag[fm], false, bfrag[fn],
                    (short)0, acc[fm][fn], false, false);
        buf ^= 1;
    }

    // ---- epilogue: bias + activation + store (C/D layout per ISA) ----
    #pragma unroll
    for (int fm = 0; fm < 2; ++fm) {
        #pragma unroll
        for (int fn = 0; fn < 4; ++fn) {
            int col = nBase + waveN * 64 + fn * 16 + l15;
            float bv = bias ? bias[col] : 0.0f;
            #pragma unroll
            for (int i = 0; i < 8; ++i) {
                int row = mBase + waveM * 32 + fm * 16 + ((lane < 16) ? i : i + 8);
                float v = acc[fm][fn][i] + bv;
                if (ACT == 1) v = fmaxf(v, 0.0f);
                if (ACT == 2) v = 1.0f / (1.0f + __expf(-v));
                if constexpr (sizeof(TO) == 2)
                    Y[(size_t)row * Nout + col] = bfval(v);
                else
                    Y[(size_t)row * Nout + col] = v;
            }
        }
    }
}

// ---------------------------------------------------------------------------
// (B,C,N) -> (B*N, C) transpose with 64x64 LDS tiles (coalesced both sides).
// grid = (N/64, C/64, B), 256 threads.
// ---------------------------------------------------------------------------
__global__ __launch_bounds__(256)
void transpose_cn_nc(const float* __restrict__ X, float* __restrict__ S)
{
    __shared__ float tile[64][65];
    int nBase = blockIdx.x * 64, cBase = blockIdx.y * 64, b = blockIdx.z;
    int t = threadIdx.x;
    #pragma unroll
    for (int i = 0; i < 16; ++i) {
        int e = i * 256 + t;
        int c = e >> 6, n = e & 63;
        tile[c][n] = X[((size_t)(b * 512 + cBase + c) << 12) + nBase + n];
    }
    __syncthreads();
    #pragma unroll
    for (int i = 0; i < 16; ++i) {
        int e = i * 256 + t;
        int n = e >> 6, c = e & 63;
        S[(size_t)((b << 12) + nBase + n) * 512 + cBase + c] = tile[c][n];
    }
}

// ---------------------------------------------------------------------------
// ctx_pre[b,h,d,e] = sum_n K[b,h,n,d] * (V*gate)[b,h,n,e]
// KV (M,1024) bf16: k = cols h*64+d, v = cols 512+h*64+e.  GATE (M,512) bf16.
// Tiles staged FEATURE-MAJOR ([d][n] / [e][n]) so both A and B frags are
// contiguous 16B runs along n -> ds_load_b128.
// grid = 32 (b*h), 128 threads = 4 waves; wave w owns d in [16w,16w+16).
// ---------------------------------------------------------------------------
__global__ __launch_bounds__(128)
void attn_ctx(const __bf16* __restrict__ KV, const __bf16* __restrict__ G,
              float* __restrict__ CTXP)
{
    __shared__ __bf16 Kt[64][40];   // [d][n]
    __shared__ __bf16 Vt[64][40];   // [e][n]  (v * gate)
    int bh = blockIdx.x; int b = bh >> 3, h = bh & 7;
    int t = threadIdx.x, lane = t & 31, wid = t >> 5;
    int khalf = lane >> 4, l15 = lane & 15;
    int dBase = wid * 16;

    v8f acc[4] = {};
    for (int nb = 0; nb < 4096; nb += 32) {
        #pragma unroll
        for (int i = 0; i < 2; ++i) {
            int g  = t + i * 128;            // 0..255 groups of 8 elems
            int r  = g >> 3;                 // n-row 0..31
            int c8 = (g & 7) << 3;           // feature offset 0..56
            size_t m = (size_t)((b << 12) + nb + r);
            v4u kq = *(const v4u*)(KV + m * 1024 + h * 64 + c8);
            v4u vq = *(const v4u*)(KV + m * 1024 + 512 + h * 64 + c8);
            v4u gq = *(const v4u*)(G + m * 512 + h * 64 + c8);
            #pragma unroll
            for (int j = 0; j < 4; ++j) {
                unsigned kw = kq[j], vw = vq[j], gw = gq[j];
                Kt[c8 + 2 * j    ][r] = __builtin_bit_cast(__bf16, (unsigned short)(kw & 0xffffu));
                Kt[c8 + 2 * j + 1][r] = __builtin_bit_cast(__bf16, (unsigned short)(kw >> 16));
                float v0 = bf2f((unsigned short)(vw & 0xffffu)) * bf2f((unsigned short)(gw & 0xffffu));
                float v1 = bf2f((unsigned short)(vw >> 16))     * bf2f((unsigned short)(gw >> 16));
                Vt[c8 + 2 * j    ][r] = bfval(v0);
                Vt[c8 + 2 * j + 1][r] = bfval(v1);
            }
        }
        __syncthreads();

        int d = dBase + l15;                       // A[d][n] rows contiguous
        v16bf a = load_frag(&Kt[d][khalf * 8], &Kt[d][16 + khalf * 8]);
        #pragma unroll
        for (int fn = 0; fn < 4; ++fn) {
            int e = fn * 16 + l15;                 // B[n][e] -> Vt[e][n] rows
            v16bf bb = load_frag(&Vt[e][khalf * 8], &Vt[e][16 + khalf * 8]);
            acc[fn] = __builtin_amdgcn_wmma_f32_16x16x32_bf16(
                false, a, false, bb, (short)0, acc[fn], false, false);
        }
        __syncthreads();
    }
    #pragma unroll
    for (int fn = 0; fn < 4; ++fn) {
        int e = fn * 16 + l15;
        #pragma unroll
        for (int i = 0; i < 8; ++i) {
            int d = dBase + ((lane < 16) ? i : i + 8);
            CTXP[(size_t)bh * 4096 + d * 64 + e] = acc[fn][i];
        }
    }
}

// softmax over d (axis=-2) of ctx[b,h,d,e] with scale = hd^-0.5 = 0.125
__global__ void ctx_softmax(float* __restrict__ C)
{
    int bh = blockIdx.x, e = threadIdx.x;         // 64 threads = e lanes
    float* p = C + (size_t)bh * 4096 + e;
    float vals[64];
    float mx = -1e30f;
    #pragma unroll
    for (int d = 0; d < 64; ++d) {
        float v = p[d * 64] * 0.125f;
        vals[d] = v;
        mx = fmaxf(mx, v);
    }
    float s = 0.0f;
    #pragma unroll
    for (int d = 0; d < 64; ++d) { vals[d] = __expf(vals[d] - mx); s += vals[d]; }
    float inv = 1.0f / s;
    #pragma unroll
    for (int d = 0; d < 64; ++d) p[d * 64] = vals[d] * inv;
}

// ---------------------------------------------------------------------------
// a[b,h,n,e] = sum_d q[b,h,n,d] * ctx[b,h,d,e];  q from u = Pb[:,512:] (bf16).
// Q staged [m][d] via async b128 copy (pitch 72); ctx (fp32) staged [e][d].
// grid = (32 m-tiles, 32 b*h), 256 threads = 8 waves; wave w owns 16 rows.
// ---------------------------------------------------------------------------
__global__ __launch_bounds__(256)
void attn_apply(const __bf16* __restrict__ P, const float* __restrict__ CTX,
                __bf16* __restrict__ A)
{
    __shared__ __bf16 Qs[128][72];   // [m][d]
    __shared__ __bf16 Cs[64][72];    // [e][d]
    int bh = blockIdx.y; int b = bh >> 3, h = bh & 7;
    int mBase = (b << 12) + blockIdx.x * 128;
    int t = threadIdx.x, lane = t & 31, wid = t >> 5;
    int khalf = lane >> 4, l15 = lane & 15;

    #pragma unroll
    for (int i = 0; i < 4; ++i) {                  // q 128x64 bf16, async copy
        int g = t + i * 256;                       // 0..1023 groups of 8
        int r = g >> 3, c8 = (g & 7) << 3;
        async_copy16(&Qs[r][c8],
                     P + (size_t)(mBase + r) * 1024 + 512 + h * 64 + c8);
    }
    #pragma unroll
    for (int i = 0; i < 4; ++i) {                  // ctx 64x64, store [e][d]
        int g = t + i * 256;
        int r = g >> 4, c4 = (g & 15) << 2;        // r = d-row, c4 = e-offset
        float4 cf = *(const float4*)(CTX + (size_t)bh * 4096 + r * 64 + c4);
        Cs[c4 + 0][r] = bfval(cf.x);
        Cs[c4 + 1][r] = bfval(cf.y);
        Cs[c4 + 2][r] = bfval(cf.z);
        Cs[c4 + 3][r] = bfval(cf.w);
    }
    wait_async0();
    __syncthreads();

    v8f acc[4] = {};
    #pragma unroll
    for (int kb = 0; kb < 64; kb += 32) {
        int m = wid * 16 + l15;
        v16bf a = load_frag(&Qs[m][kb + khalf * 8], &Qs[m][kb + 16 + khalf * 8]);
        #pragma unroll
        for (int fn = 0; fn < 4; ++fn) {
            int e = fn * 16 + l15;
            v16bf bb = load_frag(&Cs[e][kb + khalf * 8], &Cs[e][kb + 16 + khalf * 8]);
            acc[fn] = __builtin_amdgcn_wmma_f32_16x16x32_bf16(
                false, a, false, bb, (short)0, acc[fn], false, false);
        }
    }
    #pragma unroll
    for (int fn = 0; fn < 4; ++fn) {
        int e = h * 64 + fn * 16 + l15;
        #pragma unroll
        for (int i = 0; i < 8; ++i) {
            int m = mBase + wid * 16 + ((lane < 16) ? i : i + 8);
            A[(size_t)m * 512 + e] = bfval(acc[fn][i]);
        }
    }
}

// ---------------------------------------------------------------------------
// O[m,:] = LayerNorm( S[m,:] + E[m,:] ) * g + b   (row length 512, bf16 out)
// one block (256 threads, 8 waves) per row; wave32 shuffle + LDS reduction.
// ---------------------------------------------------------------------------
__global__ __launch_bounds__(256)
void ln_residual(const float* __restrict__ S, const float* __restrict__ E,
                 const float* __restrict__ g, const float* __restrict__ bt,
                 __bf16* __restrict__ O)
{
    int m = blockIdx.x, t = threadIdx.x;
    size_t base = (size_t)m * 512;
    float v0 = S[base + t] + E[base + t];
    float v1 = S[base + t + 256] + E[base + t + 256];
    float s = v0 + v1, sq = v0 * v0 + v1 * v1;
    #pragma unroll
    for (int o = 16; o > 0; o >>= 1) {
        s  += __shfl_xor(s, o, 32);
        sq += __shfl_xor(sq, o, 32);
    }
    __shared__ float sh[16];
    int wid = t >> 5, lane = t & 31;
    if (lane == 0) { sh[wid] = s; sh[wid + 8] = sq; }
    __syncthreads();
    if (t == 0) {
        float ts = 0.0f, tq = 0.0f;
        #pragma unroll
        for (int i = 0; i < 8; ++i) { ts += sh[i]; tq += sh[i + 8]; }
        float mean = ts * (1.0f / 512.0f);
        float var  = tq * (1.0f / 512.0f) - mean * mean;
        sh[0] = mean;
        sh[1] = rsqrtf(var + 1e-5f);
    }
    __syncthreads();
    float mean = sh[0], rstd = sh[1];
    O[base + t]       = bfval((v0 - mean) * rstd * g[t] + bt[t]);
    O[base + t + 256] = bfval((v1 - mean) * rstd * g[t + 256] + bt[t + 256]);
}

// depthwise 3x3, pad=1, relu(conv + bias); (m = b*4096 + y*64 + x, c); bf16 out
__global__ __launch_bounds__(256)
void dwconv3x3_relu(const float* __restrict__ T, const float* __restrict__ W,
                    const float* __restrict__ Bv, __bf16* __restrict__ O)
{
    int idx = blockIdx.x * 256 + threadIdx.x;   // M*512 threads
    int c = idx & 511;
    int m = idx >> 9;
    int b = m >> 12;
    int n = m & 4095;
    int y = n >> 6, x = n & 63;
    float s = Bv[c];
    #pragma unroll
    for (int dy = -1; dy <= 1; ++dy) {
        int yy = y + dy;
        if (yy < 0 || yy > 63) continue;
        #pragma unroll
        for (int dx = -1; dx <= 1; ++dx) {
            int xx = x + dx;
            if (xx < 0 || xx > 63) continue;
            int mm = (b << 12) + (yy << 6) + xx;
            s += T[(size_t)mm * 512 + c] * W[c * 9 + (dy + 1) * 3 + (dx + 1)];
        }
    }
    O[(size_t)m * 512 + c] = bfval(fmaxf(s, 0.0f));
}

// out[b,c,n] = (res + (t3*bn_ce_g + bn_ce_b)) * bn_out_g + bn_out_b,
// transposing (m,c) -> NCHW via 64x64 LDS tiles.  grid = (N/64, C/64, B).
__global__ __launch_bounds__(256)
void final_fuse(const float* __restrict__ RES, const float* __restrict__ T3,
                const float* __restrict__ ceg, const float* __restrict__ ceb,
                const float* __restrict__ og,  const float* __restrict__ ob,
                float* __restrict__ OUT)
{
    __shared__ float tile[64][65];
    int nBase = blockIdx.x * 64, cBase = blockIdx.y * 64, b = blockIdx.z;
    int t = threadIdx.x;
    #pragma unroll
    for (int i = 0; i < 16; ++i) {
        int e = i * 256 + t;
        int n = e >> 6, c = e & 63;
        int cc = cBase + c;
        size_t m = (size_t)((b << 12) + nBase + n);
        float r  = RES[m * 512 + cc];
        float tv = T3[m * 512 + cc] * ceg[cc] + ceb[cc];
        tile[n][c] = (r + tv) * og[cc] + ob[cc];
    }
    __syncthreads();
    #pragma unroll
    for (int i = 0; i < 16; ++i) {
        int e = i * 256 + t;
        int c = e >> 6, n = e & 63;
        OUT[((size_t)(b * 512 + cBase + c) << 12) + nBase + n] = tile[n][c];
    }
}

// ---------------------------------------------------------------------------
extern "C" void kernel_launch(void* const* d_in, const int* in_sizes, int n_in,
                              void* d_out, int out_size, void* d_ws, size_t ws_size,
                              hipStream_t stream)
{
    (void)in_sizes; (void)n_in; (void)out_size; (void)ws_size;

    const float* x1      = (const float*)d_in[0];
    const float* x2      = (const float*)d_in[1];
    const float* cp1_w   = (const float*)d_in[2];
    const float* cp1_b   = (const float*)d_in[3];
    const float* cp2_w   = (const float*)d_in[4];
    const float* cp2_b   = (const float*)d_in[5];
    const float* kv1_w   = (const float*)d_in[6];
    const float* kv2_w   = (const float*)d_in[7];
    const float* g1a_w   = (const float*)d_in[8];
    const float* g1a_b   = (const float*)d_in[9];
    const float* g1b_w   = (const float*)d_in[10];
    const float* g1b_b   = (const float*)d_in[11];
    const float* g2a_w   = (const float*)d_in[12];
    const float* g2a_b   = (const float*)d_in[13];
    const float* g2b_w   = (const float*)d_in[14];
    const float* g2b_b   = (const float*)d_in[15];
    const float* end1_w  = (const float*)d_in[16];
    const float* end1_b  = (const float*)d_in[17];
    const float* end2_w  = (const float*)d_in[18];
    const float* end2_b  = (const float*)d_in[19];
    const float* ln1_g   = (const float*)d_in[20];
    const float* ln1_b   = (const float*)d_in[21];
    const float* ln2_g   = (const float*)d_in[22];
    const float* ln2_b   = (const float*)d_in[23];
    const float* res_w   = (const float*)d_in[24];
    const float* ce1_w   = (const float*)d_in[25];
    const float* ce1_b   = (const float*)d_in[26];
    const float* dw_w    = (const float*)d_in[27];
    const float* dw_b    = (const float*)d_in[28];
    const float* ce3_w   = (const float*)d_in[29];
    const float* ce3_b   = (const float*)d_in[30];
    const float* bn_ce_g = (const float*)d_in[31];
    const float* bn_ce_b = (const float*)d_in[32];
    const float* bn_out_g= (const float*)d_in[33];
    const float* bn_out_b= (const float*)d_in[34];

    // ---- workspace layout; U = 16384*512 elements ----
    const size_t U = (size_t)16384 * 512;
    float*  ws_f   = (float*)d_ws;
    float*  S1     = ws_f;                       // U f32
    float*  S2     = ws_f + U;                   // U f32
    __bf16* P1b    = (__bf16*)(ws_f + 2 * U);    // 2U bf16
    __bf16* P2b    = P1b  + 2 * U;
    __bf16* KV1b   = P2b  + 2 * U;               // 2U bf16
    __bf16* KV2b   = KV1b + 2 * U;
    __bf16* GTb    = KV2b + 2 * U;               // U bf16
    __bf16* GATE1b = GTb    + U;
    __bf16* GATE2b = GATE1b + U;
    __bf16* A1b    = GATE2b + U;
    __bf16* A2b    = A1b    + U;
    float*  CTX1   = (float*)(A2b + U);          // 131072 f32 each
    float*  CTX2   = CTX1 + 131072;
    __bf16* WBASE  = (__bf16*)(CTX2 + 131072);   // bf16 weights (~11 MB)
    const int WN1 = 1024 * 512, WN2 = 512 * 512;
    __bf16* cp1_wb = WBASE;
    __bf16* cp2_wb = cp1_wb + WN1;
    __bf16* kv1_wb = cp2_wb + WN1;
    __bf16* kv2_wb = kv1_wb + WN1;
    __bf16* end1_wb= kv2_wb + WN1;
    __bf16* end2_wb= end1_wb + WN1;
    __bf16* res_wb = end2_wb + WN1;
    __bf16* ce1_wb = res_wb + WN1;
    __bf16* g1a_wb = ce1_wb + WN1;
    __bf16* g1b_wb = g1a_wb + WN2;
    __bf16* g2a_wb = g1b_wb + WN2;
    __bf16* g2b_wb = g2a_wb + WN2;
    __bf16* ce3_wb = g2b_wb + WN2;
    // dead-buffer reuse (stream-ordered, hazard-free):
    float*  E1  = (float*)KV1b;     // KV dead after attn_ctx
    float*  E2  = (float*)KV2b;
    __bf16* O1b = GATE1b;           // GATE dead after attn_ctx
    __bf16* O2b = GATE2b;
    float*  RES = (float*)P1b;      // P dead after end GEMMs / attn_apply
    float*  TC  = (float*)P2b;
    __bf16* TDb = GTb;              // GT dead after gate-b GEMMs
    float*  T3  = E1;               // E dead after LN

    // ---- pre-convert all weights to bf16 (L2-resident, ~11 MB) ----
    auto cvt = [&](const float* s, __bf16* d, int n) {
        f32_to_bf16<<<(n + 255) / 256, 256, 0, stream>>>(s, d, n);
    };
    cvt(cp1_w, cp1_wb, WN1);  cvt(cp2_w, cp2_wb, WN1);
    cvt(kv1_w, kv1_wb, WN1);  cvt(kv2_w, kv2_wb, WN1);
    cvt(end1_w, end1_wb, WN1); cvt(end2_w, end2_wb, WN1);
    cvt(res_w, res_wb, WN1);  cvt(ce1_w, ce1_wb, WN1);
    cvt(g1a_w, g1a_wb, WN2);  cvt(g1b_w, g1b_wb, WN2);
    cvt(g2a_w, g2a_wb, WN2);  cvt(g2b_w, g2b_wb, WN2);
    cvt(ce3_w, ce3_wb, WN2);

    dim3 tT(64, 8, 4);                 // (N/64, C/64, B)
    dim3 gWide(128, 8);                // (M/128, 1024/128)
    dim3 gNarrow(128, 4);              // (M/128,  512/128)
    dim3 gApply(32, 32);               // (m-tiles, b*h)
    const __bf16* nbf = nullptr;
    const float*  nf  = nullptr;

    // s1, s2 = NCHW -> (B*N, C)
    transpose_cn_nc<<<tT, 256, 0, stream>>>(x1, S1);
    transpose_cn_nc<<<tT, 256, 0, stream>>>(x2, S2);

    // p = relu(s @ cp_w^T + b)   (y = p[:, :512], u = p[:, 512:]); bf16 out
    gemm_wmma<1, float, __bf16><<<gWide, 256, 0, stream>>>(S1, 512, 512, nf, 0, 512, cp1_wb, cp1_b, P1b, 1024);
    gemm_wmma<1, float, __bf16><<<gWide, 256, 0, stream>>>(S2, 512, 512, nf, 0, 512, cp2_wb, cp2_b, P2b, 1024);

    // kv = u @ kv_w^T  (bf16 in, bf16 out)
    gemm_wmma<0, __bf16, __bf16><<<gWide, 256, 0, stream>>>(P1b + 512, 1024, 512, nbf, 0, 512, kv1_wb, nullptr, KV1b, 1024);
    gemm_wmma<0, __bf16, __bf16><<<gWide, 256, 0, stream>>>(P2b + 512, 1024, 512, nbf, 0, 512, kv2_wb, nullptr, KV2b, 1024);

    // gate = sigmoid( relu(u @ ga^T + b) @ gb^T + b )
    gemm_wmma<1, __bf16, __bf16><<<gNarrow, 256, 0, stream>>>(P1b + 512, 1024, 512, nbf, 0, 512, g1a_wb, g1a_b, GTb, 512);
    gemm_wmma<2, __bf16, __bf16><<<gNarrow, 256, 0, stream>>>(GTb, 512, 512, nbf, 0, 512, g1b_wb, g1b_b, GATE1b, 512);
    gemm_wmma<1, __bf16, __bf16><<<gNarrow, 256, 0, stream>>>(P2b + 512, 1024, 512, nbf, 0, 512, g2a_wb, g2a_b, GTb, 512);
    gemm_wmma<2, __bf16, __bf16><<<gNarrow, 256, 0, stream>>>(GTb, 512, 512, nbf, 0, 512, g2b_wb, g2b_b, GATE2b, 512);

    // ctx = softmax( scale * K^T (V*gate) )   (fp32 ctx)
    attn_ctx<<<32, 128, 0, stream>>>(KV1b, GATE1b, CTX1);
    attn_ctx<<<32, 128, 0, stream>>>(KV2b, GATE2b, CTX2);
    ctx_softmax<<<32, 64, 0, stream>>>(CTX1);
    ctx_softmax<<<32, 64, 0, stream>>>(CTX2);

    // a1 = q1 @ ctx2, a2 = q2 @ ctx1   (cross-stream; bf16 out)
    attn_apply<<<gApply, 256, 0, stream>>>(P1b, CTX2, A1b);
    attn_apply<<<gApply, 256, 0, stream>>>(P2b, CTX1, A2b);

    // e = concat([y, a]) @ end_w^T + b   (virtual concat; fp32 out for LN)
    gemm_wmma<0, __bf16, float><<<gNarrow, 256, 0, stream>>>(P1b, 1024, 512, A1b, 512, 1024, end1_wb, end1_b, E1, 512);
    gemm_wmma<0, __bf16, float><<<gNarrow, 256, 0, stream>>>(P2b, 1024, 512, A2b, 512, 1024, end2_wb, end2_b, E2, 512);

    // o = LN(s + e)  (bf16 out)
    ln_residual<<<16384, 256, 0, stream>>>(S1, E1, ln1_g, ln1_b, O1b);
    ln_residual<<<16384, 256, 0, stream>>>(S2, E2, ln2_g, ln2_b, O2b);

    // 1x1 convs on fused = concat([o1, o2]) (virtual concat via split-K)
    gemm_wmma<0, __bf16, float><<<gNarrow, 256, 0, stream>>>(O1b, 512, 512, O2b, 512, 1024, res_wb, nullptr, RES, 512);
    gemm_wmma<0, __bf16, float><<<gNarrow, 256, 0, stream>>>(O1b, 512, 512, O2b, 512, 1024, ce1_wb, ce1_b, TC, 512);

    // relu(depthwise 3x3 + bias)  (bf16 out)
    dwconv3x3_relu<<<32768, 256, 0, stream>>>(TC, dw_w, dw_b, TDb);

    // ce3 1x1 + bias (fp32 out)
    gemm_wmma<0, __bf16, float><<<gNarrow, 256, 0, stream>>>(TDb, 512, 512, nbf, 0, 512, ce3_wb, ce3_b, T3, 512);

    // out = (res + (t*bn_ce_g + bn_ce_b)) * bn_out_g + bn_out_b  -> NCHW
    final_fuse<<<tT, 256, 0, stream>>>(RES, T3, bn_ce_g, bn_ce_b,
                                       bn_out_g, bn_out_b, (float*)d_out);
}
